// SNN_Delay_Conv_54107997995105
// MI455X (gfx1250) — matrix operand order repaired and checked by reading, not compiled, not run on hardware
//
#include <hip/hip_runtime.h>
#include <hip/hip_bf16.h>

typedef __attribute__((ext_vector_type(16))) _Float16 v16h;
typedef __attribute__((ext_vector_type(8)))  float    v8f;

#define DEV __device__ __forceinline__

// ---------------------------------------------------------------------------
// Tap builder: taps[k][o] (f16, k-major) = W[o,i,s] * gauss(P[o,i])[d],
// k = (i*S + s)*5 + d.  S = 25 for conv layers, 1 for FC layers.
// ---------------------------------------------------------------------------
__global__ void k_build_taps(const float* __restrict__ W, const float* __restrict__ P,
                             _Float16* __restrict__ taps, int O, int I, int S) {
  int oi = blockIdx.x * blockDim.x + threadIdx.x;
  if (oi >= O * I) return;
  int o = oi / I, i = oi % I;
  float p = P[oi];
  float g[5];
  float s = 0.f;
#pragma unroll
  for (int d = 0; d < 5; ++d) {
    float xx = ((float)d - (2.0f + p)) * (1.0f / (2.0f + 1e-7f));
    g[d] = expf(-0.5f * xx * xx);
    s += g[d];
  }
  float inv = 1.0f / (s + 1e-7f);
  for (int sp = 0; sp < S; ++sp) {
    float w = W[(o * I + i) * S + sp];
#pragma unroll
    for (int d = 0; d < 5; ++d)
      taps[((i * S + sp) * 5 + d) * O + o] = (_Float16)(w * g[d] * inv);
  }
}

// ---------------------------------------------------------------------------
// Input transpose: (B,T=50,2,34,34) f32 -> (B,2,34,34,50) f16 (t innermost so
// A-fragment gathers are lane-coalesced).
// ---------------------------------------------------------------------------
__global__ void k_xpose(const float* __restrict__ in, _Float16* __restrict__ out, int n) {
  int i = blockIdx.x * blockDim.x + threadIdx.x;   // i = (b*2+c)*1156 + hw
  if (i >= n) return;
  int hw = i % 1156; int r = i / 1156; int c = r % 2; int b = r / 2;
  for (int t = 0; t < 50; ++t)
    out[(size_t)i * 50 + t] = (_Float16)in[((size_t)(b * 50 + t) * 2 + c) * 1156 + hw];
}

// ---------------------------------------------------------------------------
// Layer problem descriptors.  x is a *guarded* pointer (buffer has a 16-elem
// front guard, so rowBase-5 underruns are safe unconditional loads).
// A offset table entry: (d << 26) | (spatial_offset + d); sentinel d=31 for
// zero-padded k.  addr = x[rowBase + off] with rowBase including (t - 5).
// Output index = outBase(m) + n*NStride.
// ---------------------------------------------------------------------------
struct Conv1P {  // out (B,12,30,30,51); K = 2*5*5*5 = 250
  const _Float16* x; const _Float16* taps; const float* bias; float* out;
  static constexpr int Kreal = 250, Kpad = 256, KSEC = 256, Tin = 50, Nreal = 12;
  static constexpr int NStride = 45900;  // 30*30*51
  static DEV int tabEntry(int k) {
    if (k >= Kreal) return 31 << 26;
    int d = k % 5; int k2 = k / 5; int kw = k2 % 5; int kh = (k2 / 5) % 5; int i = k2 / 25;
    return (d << 26) | (i * 57800 + kh * 1700 + kw * 50 + d);
  }
  static DEV int rowT(int m) { return m % 51; }
  static DEV int rowBase(int m) {
    int t = m % 51; int r = m / 51; int wo = r % 30; r /= 30; int ho = r % 30; int b = r / 30;
    return b * 115600 + ho * 1700 + wo * 50 + t - 5;
  }
  static DEV int outBase(int m) {
    int t = m % 51; int r = m / 51; int wo = r % 30; r /= 30; int ho = r % 30; int b = r / 30;
    return b * 550800 + ho * 1530 + wo * 51 + t;
  }
};

struct Conv2P {  // x spikes1 (B,12,15,15,51); out (B,32,11,11,52); K=1500
  const _Float16* x; const _Float16* taps; const float* bias; float* out;
  static constexpr int Kreal = 1500, Kpad = 1536, KSEC = 512, Tin = 51, Nreal = 32;
  static constexpr int NStride = 6292;  // 11*11*52
  static DEV int tabEntry(int k) {
    if (k >= Kreal) return 31 << 26;
    int d = k % 5; int k2 = k / 5; int kw = k2 % 5; int kh = (k2 / 5) % 5; int i = k2 / 25;
    return (d << 26) | (i * 11475 + kh * 765 + kw * 51 + d);
  }
  static DEV int rowT(int m) { return m % 52; }
  static DEV int rowBase(int m) {
    int t = m % 52; int r = m / 52; int wo = r % 11; int ho = (r / 11) % 11; int b = r / 121;
    return b * 137700 + ho * 765 + wo * 51 + t - 5;
  }
  static DEV int outBase(int m) {
    int t = m % 52; int r = m / 52; int wo = r % 11; int ho = (r / 11) % 11; int b = r / 121;
    return b * 201344 + ho * 572 + wo * 52 + t;
  }
};

struct FC3P {  // x spikes2 (B,800,52); out (B,100,53); K=4000
  const _Float16* x; const _Float16* taps; const float* bias; float* out;
  static constexpr int Kreal = 4000, Kpad = 4096, KSEC = 512, Tin = 52, Nreal = 100;
  static constexpr int NStride = 53;
  static DEV int tabEntry(int k) {
    if (k >= Kreal) return 31 << 26;
    int d = k % 5; int i = k / 5;
    return (d << 26) | (i * 52 + d);
  }
  static DEV int rowT(int m) { return m % 53; }
  static DEV int rowBase(int m) { return (m / 53) * 41600 + (m % 53) - 5; }
  static DEV int outBase(int m) { return (m / 53) * 5300 + (m % 53); }
};

struct FC4P {  // x spikes3 (B,100,53); out (B,10,54); K=500
  const _Float16* x; const _Float16* taps; const float* bias; float* out;
  static constexpr int Kreal = 500, Kpad = 512, KSEC = 512, Tin = 53, Nreal = 10;
  static constexpr int NStride = 54;
  static DEV int tabEntry(int k) {
    if (k >= Kreal) return 31 << 26;
    int d = k % 5; int i = k / 5;
    return (d << 26) | (i * 53 + d);
  }
  static DEV int rowT(int m) { return m % 54; }
  static DEV int rowBase(int m) { return (m / 54) * 5300 + (m % 54) - 5; }
  static DEV int outBase(int m) { return (m / 54) * 540 + (m % 54); }
};

// ---------------------------------------------------------------------------
// WMMA implicit-GEMM.  Block = 8 waves sharing one N-tile; B panel staged in
// LDS (n-major: B fragment = one 32-byte contiguous LDS read); A offset table
// in LDS removes all per-element div/mod; A loads are unconditional guarded
// global_load_u16 (lane-coalesced) + cndmask.  C tile is transposed through
// LDS so the epilogue decodes ONE output address per lane.
// ---------------------------------------------------------------------------
template <class P>
__global__ void __launch_bounds__(256) k_gemm(P p, int mblocks) {
  __shared__ int tab[P::Kpad];
  __shared__ alignas(32) _Float16 bs[16][P::KSEC];
  __shared__ float cs[8][16][17];

  const int tid   = threadIdx.x;
  const int ntile = blockIdx.x / mblocks;
  const int mb    = blockIdx.x % mblocks;
  const int lane  = tid & 31;
  const int wv    = tid >> 5;
  const int hi    = (lane >> 4) & 1;
  const int mtile = mb * 8 + wv;
  const int nloc  = lane & 15;
  const int nbase = ntile * 16;
  const int mrow  = mtile * 16 + (lane & 15);

  const int trow = P::rowT(mrow);
  const _Float16* __restrict__ xr = p.x + P::rowBase(mrow);
  __builtin_prefetch(xr, 0, 1);

  // bias values this lane will need in the epilogue (clamped: no OOB)
  float biasv[8];
#pragma unroll
  for (int j = 0; j < 8; ++j) {
    int n = nbase + hi * 8 + j;
    biasv[j] = p.bias[n < P::Nreal ? n : P::Nreal - 1];
  }

  for (int k = tid; k < P::Kpad; k += 256) tab[k] = P::tabEntry(k);

  v8f acc = {};
  constexpr int sections = (P::Kpad + P::KSEC - 1) / P::KSEC;
  for (int sec = 0; sec < sections; ++sec) {
    const int kbase = sec * P::KSEC;
    __syncthreads();
    // stage zero-padded B panel for this k-section
    for (int idx = tid; idx < 16 * P::KSEC; idx += 256) {
      const int kk = idx & (P::KSEC - 1);
      const int n  = idx / P::KSEC;
      const int k  = kbase + kk;
      _Float16 v = (_Float16)0.f;
      if (k < P::Kreal && (nbase + n) < P::Nreal) v = p.taps[k * P::Nreal + nbase + n];
      bs[n][kk] = v;
    }
    __syncthreads();
    const int rem = P::Kpad - kbase;
    const int kcs = (rem < P::KSEC ? rem : P::KSEC) >> 5;
    for (int kc = 0; kc < kcs; ++kc) {
      // B fragment: k = kbase + kc*32 + 16*hi + e  (contiguous 32B in LDS)
      v16h b = *(const v16h*)&bs[nloc][kc * 32 + hi * 16];
      // A fragment: k = kA + e
      v16h a;
      const int kA = kbase + kc * 32 + hi * 16;
#pragma unroll
      for (int e = 0; e < 16; ++e) {
        const int k = kA + e;
        const unsigned te = (unsigned)tab[k];
        const int d   = (int)(te >> 26);
        const int off = (int)(te & 0x03ffffffu);
        _Float16 v = xr[off];  // guarded, always safe
        a[e] = ((unsigned)(trow + d - 5) < (unsigned)P::Tin) ? v : (_Float16)0.f;
      }
      acc = __builtin_amdgcn_wmma_f32_16x16x32_f16(false, a, false, b, (short)0,
                                                   acc, false, false);
    }
  }

  // ---- epilogue: transpose C tile through LDS, one address decode per lane
#pragma unroll
  for (int v = 0; v < 8; ++v) cs[wv][v + hi * 8][nloc] = acc[v];
  __syncthreads();
  const int row = lane & 15;
  const int ob  = P::outBase(mtile * 16 + row);  // same m as this lane's A row
#pragma unroll
  for (int j = 0; j < 8; ++j) {
    const int n = nbase + hi * 8 + j;
    if (n < P::Nreal)
      p.out[ob + n * P::NStride] = cs[wv][row][hi * 8 + j] + biasv[j];
  }
}

// ---------------------------------------------------------------------------
// Fused 2x2 maxpool + LayerNorm(C,Hp,Wp) per (b,t).  One block per (b,t).
// ---------------------------------------------------------------------------
__global__ void __launch_bounds__(256)
k_pool_ln(const float* __restrict__ in, float* __restrict__ out,
          int C, int Hc, int Wc, int T) {
  const int Hp = Hc / 2, Wp = Wc / 2;
  const int t = blockIdx.x % T, b = blockIdx.x / T;
  const int N = C * Hp * Wp;
  __shared__ float s1[256], s2[256];
  float sum = 0.f, sq = 0.f;
  for (int idx = threadIdx.x; idx < N; idx += blockDim.x) {
    int wp = idx % Wp; int r = idx / Wp; int hp = r % Hp; int c = r / Hp;
    const float* base = in + ((((size_t)b * C + c) * Hc + 2 * hp) * Wc + 2 * wp) * T + t;
    float mx = fmaxf(fmaxf(base[0], base[(size_t)T]),
                     fmaxf(base[(size_t)Wc * T], base[(size_t)Wc * T + T]));
    sum += mx; sq += mx * mx;
    out[((((size_t)b * C + c) * Hp + hp) * Wp + wp) * T + t] = mx;
  }
  s1[threadIdx.x] = sum; s2[threadIdx.x] = sq;
  __syncthreads();
  for (int ofs = 128; ofs > 0; ofs >>= 1) {
    if ((int)threadIdx.x < ofs) {
      s1[threadIdx.x] += s1[threadIdx.x + ofs];
      s2[threadIdx.x] += s2[threadIdx.x + ofs];
    }
    __syncthreads();
  }
  float mean = s1[0] / (float)N;
  float var  = s2[0] / (float)N - mean * mean;
  float rs   = rsqrtf(var + 1e-5f);
  for (int idx = threadIdx.x; idx < N; idx += blockDim.x) {
    int wp = idx % Wp; int r = idx / Wp; int hp = r % Hp; int c = r / Hp;
    size_t pos = ((((size_t)b * C + c) * Hp + hp) * Wp + wp) * T + t;
    out[pos] = (out[pos] - mean) * rs;
  }
}

// ---------------------------------------------------------------------------
// LIF scan (reset-by-subtract, detached reset).  One thread per neuron.
// ---------------------------------------------------------------------------
__global__ void k_lif(const float* __restrict__ x, _Float16* __restrict__ spk,
                      int nNeur, int T) {
  int n = blockIdx.x * blockDim.x + threadIdx.x;
  if (n >= nNeur) return;
  const float* xr = x + (size_t)n * T;
  _Float16* sr = spk + (size_t)n * T;
  float mem = 0.f, sp = 0.f;
  for (int t = 0; t < T; ++t) {
    mem = 0.9f * mem + xr[t] - sp * 1.0f;
    sp = (mem - 1.0f) > 0.f ? 1.f : 0.f;
    sr[t] = (_Float16)sp;
  }
}

// Final LIF: writes spikes and membranes to d_out in (T,B,O) layout.
__global__ void k_lif4(const float* __restrict__ x, float* __restrict__ out_spk,
                       float* __restrict__ out_mem, int Bn, int O, int T) {
  int n = blockIdx.x * blockDim.x + threadIdx.x;
  if (n >= Bn * O) return;
  int b = n / O, o = n % O;
  const float* xr = x + (size_t)n * T;
  float mem = 0.f, sp = 0.f;
  for (int t = 0; t < T; ++t) {
    mem = 0.9f * mem + xr[t] - sp * 1.0f;
    sp = (mem - 1.0f) > 0.f ? 1.f : 0.f;
    out_spk[(size_t)t * Bn * O + b * O + o] = sp;
    out_mem[(size_t)t * Bn * O + b * O + o] = mem;
  }
}

// ---------------------------------------------------------------------------
// Host launcher
// ---------------------------------------------------------------------------
extern "C" void kernel_launch(void* const* d_in, const int* in_sizes, int n_in,
                              void* d_out, int out_size, void* d_ws, size_t ws_size,
                              hipStream_t stream) {
  (void)in_sizes; (void)n_in; (void)out_size; (void)ws_size;
  const float* data = (const float*)d_in[0];
  const float* w1 = (const float*)d_in[1];  const float* b1 = (const float*)d_in[2];
  const float* p1 = (const float*)d_in[3];
  const float* w2 = (const float*)d_in[4];  const float* b2 = (const float*)d_in[5];
  const float* p2 = (const float*)d_in[6];
  const float* w3 = (const float*)d_in[7];  const float* b3 = (const float*)d_in[8];
  const float* p3 = (const float*)d_in[9];
  const float* w4 = (const float*)d_in[10]; const float* b4 = (const float*)d_in[11];
  const float* p4 = (const float*)d_in[12];

  char* base = (char*)d_ws;
  size_t off = 0;
  auto alloc = [&](size_t bytes) {
    size_t o = off;
    off = (off + bytes + 255) & ~(size_t)255;
    return o;
  };
  // f16 buffers that feed A fragments get a 16-element (32B) front guard.
  const size_t o_t1 = alloc((size_t)250 * 12 * 2);
  const size_t o_t2 = alloc((size_t)1500 * 32 * 2);
  const size_t o_t3 = alloc((size_t)4000 * 100 * 2);
  const size_t o_t4 = alloc((size_t)500 * 10 * 2);
  const size_t o_xt = alloc(((size_t)128 * 2 * 34 * 34 * 50 + 16) * 2);
  const size_t o_c1 = alloc((size_t)128 * 12 * 30 * 30 * 51 * 4);
  const size_t o_l1 = alloc((size_t)128 * 12 * 15 * 15 * 51 * 4);
  const size_t o_s1 = alloc(((size_t)128 * 12 * 15 * 15 * 51 + 16) * 2);
  const size_t o_c2 = alloc((size_t)128 * 32 * 11 * 11 * 52 * 4);
  const size_t o_l2 = alloc((size_t)128 * 32 * 5 * 5 * 52 * 4);
  const size_t o_s2 = alloc(((size_t)128 * 32 * 5 * 5 * 52 + 16) * 2);
  const size_t o_f3 = alloc((size_t)128 * 100 * 53 * 4);
  const size_t o_s3 = alloc(((size_t)128 * 100 * 53 + 16) * 2);
  const size_t o_f4 = alloc((size_t)128 * 10 * 54 * 4);

  _Float16* taps1 = (_Float16*)(base + o_t1);
  _Float16* taps2 = (_Float16*)(base + o_t2);
  _Float16* taps3 = (_Float16*)(base + o_t3);
  _Float16* taps4 = (_Float16*)(base + o_t4);
  _Float16* xt    = (_Float16*)(base + o_xt) + 16;  // guarded
  float*    c1    = (float*)(base + o_c1);
  float*    l1    = (float*)(base + o_l1);
  _Float16* s1    = (_Float16*)(base + o_s1) + 16;  // guarded
  float*    c2    = (float*)(base + o_c2);
  float*    l2    = (float*)(base + o_l2);
  _Float16* s2    = (_Float16*)(base + o_s2) + 16;  // guarded
  float*    f3    = (float*)(base + o_f3);
  _Float16* s3    = (_Float16*)(base + o_s3) + 16;  // guarded
  float*    f4    = (float*)(base + o_f4);

  auto cdiv = [](int a, int b) { return (a + b - 1) / b; };

  // 1) delay-weighted taps + input transpose
  k_build_taps<<<cdiv(12 * 2, 256),    256, 0, stream>>>(w1, p1, taps1, 12, 2, 25);
  k_build_taps<<<cdiv(32 * 12, 256),   256, 0, stream>>>(w2, p2, taps2, 32, 12, 25);
  k_build_taps<<<cdiv(100 * 800, 256), 256, 0, stream>>>(w3, p3, taps3, 100, 800, 1);
  k_build_taps<<<cdiv(10 * 100, 256),  256, 0, stream>>>(w4, p4, taps4, 10, 100, 1);
  k_xpose<<<cdiv(128 * 2 * 34 * 34, 256), 256, 0, stream>>>(data, xt, 128 * 2 * 34 * 34);

  // 2) conv1 (M = 128*30*30*51 = 5875200, N=12->16, K=250->256)
  {
    Conv1P p{xt, taps1, b1, c1};
    int mblocks = (128 * 30 * 30 * 51) / 16 / 8;  // 45900, exact
    k_gemm<Conv1P><<<1 * mblocks, 256, 0, stream>>>(p, mblocks);
  }
  k_pool_ln<<<128 * 51, 256, 0, stream>>>(c1, l1, 12, 30, 30, 51);
  k_lif<<<cdiv(128 * 12 * 15 * 15, 256), 256, 0, stream>>>(l1, s1, 128 * 12 * 15 * 15, 51);

  // 3) conv2 (M = 128*11*11*52 = 805376, N=32, K=1500->1536)
  {
    Conv2P p{s1, taps2, b2, c2};
    int mblocks = (128 * 11 * 11 * 52) / 16 / 8;  // 6292, exact
    k_gemm<Conv2P><<<2 * mblocks, 256, 0, stream>>>(p, mblocks);
  }
  k_pool_ln<<<128 * 52, 256, 0, stream>>>(c2, l2, 32, 11, 11, 52);
  k_lif<<<cdiv(128 * 32 * 5 * 5, 256), 256, 0, stream>>>(l2, s2, 128 * 32 * 5 * 5, 52);

  // 4) fc3 (M = 128*53 = 6784, N=100->112, K=4000->4096)
  {
    FC3P p{s2, taps3, b3, f3};
    int mblocks = (128 * 53) / 16 / 8;  // 53, exact
    k_gemm<FC3P><<<7 * mblocks, 256, 0, stream>>>(p, mblocks);
  }
  k_lif<<<cdiv(128 * 100, 256), 256, 0, stream>>>(f3, s3, 128 * 100, 53);

  // 5) fc4 (M = 128*54 = 6912, N=10->16, K=500->512)
  {
    FC4P p{s3, taps4, b4, f4};
    int mblocks = (128 * 54) / 16 / 8;  // 54, exact
    k_gemm<FC4P><<<1 * mblocks, 256, 0, stream>>>(p, mblocks);
  }
  float* out = (float*)d_out;
  k_lif4<<<cdiv(128 * 10, 256), 256, 0, stream>>>(f4, out, out + 54 * 128 * 10, 128, 10, 54);
}
